// ResNetBlock_43885975830722
// MI455X (gfx1250) — matrix operand order repaired and checked by reading, not compiled, not run on hardware
//
#include <hip/hip_runtime.h>

typedef __attribute__((ext_vector_type(16))) __bf16        v16bf;
typedef __attribute__((ext_vector_type(8)))  float         v8f;
typedef __attribute__((ext_vector_type(8)))  unsigned int  v8u;

#define EPS_BN 1e-3f

// ---------- bf16 helpers (round-to-nearest-even) ----------
__device__ __forceinline__ unsigned int f2bf1(float x) {
    unsigned int u = __float_as_uint(x);
    return (u + 0x7FFFu + ((u >> 16) & 1u)) >> 16;
}
__device__ __forceinline__ unsigned int packbf(float lo, float hi) {
    return f2bf1(lo) | (f2bf1(hi) << 16);
}

// ---------------------------------------------------------------------------
// Pre-pass: repack W [P=4][A=8][O=64][C=64] f32 into bf16 WMMA-B-operand
// layout.  BW index:  ((((p*8+a)*4 + nt)*2 + half)*32 + lane)*8 + g
// B (32x16, bf16): lane j holds N=j&15; VGPR g holds K = ((g>>2)<<4) +
// ((j>>4)<<3) + 2(g&3) (lo) and that + 1 (hi).  c = half*32 + K.
// ---------------------------------------------------------------------------
__global__ __launch_bounds__(256) void pack_weights(
    const float* __restrict__ W, unsigned int* __restrict__ BW)
{
    int t    = blockIdx.x * 256 + threadIdx.x;   // 0 .. 65535
    int g    = t & 7;
    int lane = (t >> 3)  & 31;
    int half = (t >> 8)  & 1;
    int nt   = (t >> 9)  & 3;
    int a    = (t >> 11) & 7;
    int p    = (t >> 14) & 3;
    int o    = nt * 16 + (lane & 15);
    int c    = half * 32 + ((g >> 2) << 4) + ((lane >> 4) << 3) + ((g & 3) << 1);
    const float* row = W + ((p * 8 + a) * 64 + o) * 64;
    BW[t] = packbf(row[c], row[c + 1]);
}

// ---------------------------------------------------------------------------
// One geoconv layer: interp -> (per-rotation) GEMM via WMMA -> bias/ReLU ->
// rotation max -> BatchNorm (+ optional residual).
// Block = 16 vertices, 128 threads = 4 wave32 (wave r handles rotation r).
// ---------------------------------------------------------------------------
__global__ __launch_bounds__(128) void geoconv(
    const float* __restrict__ sig,      // gather source [V,64]
    const int*   __restrict__ bc_idx,   // [V,4,8,3]
    const float* __restrict__ bc_w,     // [V,4,8,3]
    const unsigned int* __restrict__ BW,// packed bf16 weights (B layout)
    const float* __restrict__ bias,
    const float* __restrict__ gmm, const float* __restrict__ bet,
    const float* __restrict__ mea, const float* __restrict__ var,
    const float* __restrict__ resid,    // nullptr for layer 1
    float* __restrict__ out)
{
    // 64 K-chunks x (32 lanes x 8 dwords) of bf16 interp in WMMA-A layout.
    __shared__ __align__(32) unsigned int ldsA[16384];   // 64 KB

    const int tid = threadIdx.x;
    const int v0  = blockIdx.x * 16;

    // ---------------- Phase 1: barycentric interp -> LDS (bf16, A layout) --
    // A (16x32, bf16): lane = m + ((K>>3)&1)*16 ; g = ((K>>4)<<2)|((K>>1)&3)
    #pragma unroll
    for (int it = 0; it < 4; ++it) {
        int t  = tid + it * 128;        // 512 (m,p,a) triples
        int m  = t >> 5;
        int pa = t & 31;
        int v  = v0 + m;
        int base = (v * 32 + pa) * 3;
        int i0 = bc_idx[base + 0], i1 = bc_idx[base + 1], i2 = bc_idx[base + 2];
        float w0 = bc_w[base + 0], w1 = bc_w[base + 1], w2 = bc_w[base + 2];
        const float2* s0 = (const float2*)(sig + i0 * 64);
        const float2* s1 = (const float2*)(sig + i1 * 64);
        const float2* s2 = (const float2*)(sig + i2 * 64);
        #pragma unroll
        for (int ch = 0; ch < 32; ++ch) {
            float2 a0 = s0[ch], a1 = s1[ch], a2 = s2[ch];
            float lo = w0 * a0.x + w1 * a1.x + w2 * a2.x;
            float hi = w0 * a0.y + w1 * a1.y + w2 * a2.y;
            int c    = ch * 2;
            int kc   = pa * 2 + (c >> 5);
            int k32  = c & 31;
            int lane = m + ((k32 >> 3) & 1) * 16;
            int g    = ((k32 >> 4) << 2) | ((k32 >> 1) & 3);
            ldsA[kc * 256 + lane * 8 + g] = packbf(lo, hi);
        }
    }
    __syncthreads();

    // ---------------- Phase 2: per-rotation GEMM (bf16 WMMA, f32 accum) ----
    const int wave  = tid >> 5;       // rotation index r
    const int lane  = tid & 31;
    const int shift = wave * 2;       // DELTA = 2

    v8f acc[4] = {};                  // 4 N-tiles of 16  (N = 64 per rotation)

    for (int p = 0; p < 4; ++p) {
        for (int ap = 0; ap < 8; ++ap) {           // weight angular index a'
            int a   = (ap + shift) & 7;            // interp angular index
            int paA = p * 8 + a;
            int paW = p * 8 + ap;
            #pragma unroll
            for (int half = 0; half < 2; ++half) { // c-chunks of 32
                v8u araw = *(const v8u*)&ldsA[(paA * 2 + half) * 256 + lane * 8];
                v16bf Am = __builtin_bit_cast(v16bf, araw);
                #pragma unroll
                for (int nt = 0; nt < 4; ++nt) {
                    v8u braw = *(const v8u*)(
                        BW + ((((paW * 4 + nt) * 2 + half)) << 8) + lane * 8);
                    v16bf Bm = __builtin_bit_cast(v16bf, braw);
                    acc[nt] = __builtin_amdgcn_wmma_f32_16x16x32_bf16(
                        false, Am, false, Bm, (short)0, acc[nt], false, false);
                }
            }
        }
    }

    // ---------------- Phase 3: bias + ReLU, stage per-rotation to LDS ------
    __syncthreads();                       // all interp reads done
    float* outbuf = (float*)ldsA;          // reuse 16 KB: [r][16m][64o]
    #pragma unroll
    for (int nt = 0; nt < 4; ++nt) {
        #pragma unroll
        for (int i = 0; i < 8; ++i) {
            int m = i + ((lane >> 4) << 3);
            int o = nt * 16 + (lane & 15);
            float d = acc[nt][i] + bias[o];
            outbuf[wave * 1024 + m * 64 + o] = fmaxf(d, 0.0f);
        }
    }
    __syncthreads();

    // ---------------- Phase 4: rotation max + BN (+ residual), store -------
    #pragma unroll
    for (int i = 0; i < 8; ++i) {
        int e = tid + i * 128;             // 0 .. 1023 = m*64 + o
        float x = outbuf[e];
        x = fmaxf(x, outbuf[1024 + e]);
        x = fmaxf(x, outbuf[2048 + e]);
        x = fmaxf(x, outbuf[3072 + e]);
        int m = e >> 6;
        int o = e & 63;
        float y = gmm[o] * (x - mea[o]) * rsqrtf(var[o] + EPS_BN) + bet[o];
        int v = v0 + m;
        if (resid) y += resid[v * 64 + o];
        out[v * 64 + o] = y;
    }
}

// ---------------------------------------------------------------------------
extern "C" void kernel_launch(void* const* d_in, const int* in_sizes, int n_in,
                              void* d_out, int out_size, void* d_ws, size_t ws_size,
                              hipStream_t stream) {
    const float* signal = (const float*)d_in[0];
    const int*   bc_idx = (const int*)  d_in[1];
    const float* bc_w   = (const float*)d_in[2];
    const float* W1  = (const float*)d_in[3];
    const float* b1  = (const float*)d_in[4];
    const float* g1  = (const float*)d_in[5];
    const float* be1 = (const float*)d_in[6];
    const float* m1  = (const float*)d_in[7];
    const float* v1  = (const float*)d_in[8];
    const float* W2  = (const float*)d_in[9];
    const float* b2  = (const float*)d_in[10];
    const float* g2  = (const float*)d_in[11];
    const float* be2 = (const float*)d_in[12];
    const float* m2  = (const float*)d_in[13];
    const float* v2  = (const float*)d_in[14];

    unsigned int* BW1 = (unsigned int*)d_ws;       // 256 KB
    unsigned int* BW2 = BW1 + 65536;               // 256 KB
    float*        x1  = (float*)(BW2 + 65536);     // 30000*64 f32 = 7.68 MB
    float*        outp = (float*)d_out;

    pack_weights<<<256, 256, 0, stream>>>(W1, BW1);
    pack_weights<<<256, 256, 0, stream>>>(W2, BW2);

    // Layer 1: geoconv(signal) -> x1   (no residual)
    geoconv<<<1875, 128, 0, stream>>>(signal, bc_idx, bc_w, BW1,
                                      b1, g1, be1, m1, v1, nullptr, x1);
    // Layer 2: geoconv(x1) + signal -> out
    geoconv<<<1875, 128, 0, stream>>>(x1, bc_idx, bc_w, BW2,
                                      b2, g2, be2, m2, v2, signal, outp);
}
